// MultiSelfAttention_62955630625281
// MI455X (gfx1250) — compile-verified, matrix operand order checked by reading, not run on hardware
//
#include <hip/hip_runtime.h>
#include <hip/hip_bf16.h>

typedef __attribute__((ext_vector_type(16))) _Float16 v16h;
typedef __attribute__((ext_vector_type(8)))  _Float16 v8h;
typedef __attribute__((ext_vector_type(4)))  _Float16 v4h;
typedef __attribute__((ext_vector_type(8)))  float    v8f;

#define EMB 1024
#define SEQ 2048
#define HEADS 16
#define HDIM 64
#define BATCH 2
#define MTOT (BATCH * SEQ)   // 4096 rows

// One wave per block; min-waves-per-EU=1 lifts the VGPR cap so the
// double-buffered fragment set stays in registers (no scratch spills).
#define WAVE_KERNEL __global__ __launch_bounds__(32, 1)

// ---------------------------------------------------------------------------
// Fragment loader for v_wmma_f32_16x16x32_f16 A/B operands.
// ISA 16-bit 16x32 layout: lane L holds row (L%16); K chunks:
//   lanes 0-15 : K = 0..7 and 16..23
//   lanes 16-31: K = 8..15 and 24..31
// -> two contiguous 16-byte chunks at col0+8*(L/16) and +16 more.
// ---------------------------------------------------------------------------
__device__ __forceinline__ v16h load_frag(const _Float16* __restrict__ base,
                                          int stride, int row0, int col0,
                                          int lane) {
  const _Float16* p = base + (size_t)(row0 + (lane & 15)) * stride
                           + col0 + ((lane >> 4) << 3);
  union { v16h v; v8h h[2]; } u;
  u.h[0] = *(const v8h*)(p);
  u.h[1] = *(const v8h*)(p + 16);
  return u.v;
}

__device__ __forceinline__ v8f wmma_f16(v16h a, v16h b, v8f c) {
  return __builtin_amdgcn_wmma_f32_16x16x32_f16(false, a, false, b,
                                                (short)0, c, false, false);
}

__device__ __forceinline__ float rmax16(float v) {
  v = fmaxf(v, __shfl_xor(v, 1, 16));
  v = fmaxf(v, __shfl_xor(v, 2, 16));
  v = fmaxf(v, __shfl_xor(v, 4, 16));
  v = fmaxf(v, __shfl_xor(v, 8, 16));
  return v;
}
__device__ __forceinline__ float rsum16(float v) {
  v += __shfl_xor(v, 1, 16);
  v += __shfl_xor(v, 2, 16);
  v += __shfl_xor(v, 4, 16);
  v += __shfl_xor(v, 8, 16);
  return v;
}

// ---------------------------------------------------------------------------
// Stage 0: f32 -> f16 conversion, vectorized x4
// ---------------------------------------------------------------------------
__global__ void cvt_f32_f16(const float* __restrict__ in,
                            _Float16* __restrict__ out, int n4) {
  int i = blockIdx.x * blockDim.x + threadIdx.x;
  if (i < n4) {
    float4 f = ((const float4*)in)[i];
    v4h o = { (_Float16)f.x, (_Float16)f.y, (_Float16)f.z, (_Float16)f.w };
    ((v4h*)out)[i] = o;
  }
}

// ---------------------------------------------------------------------------
// Shared GEMM core: acc[8] = A[32 rows] @ W^T[64 cols], double-buffered
// k-loop: prefetch k+32 fragments before the 8 WMMAs of the current step so
// the matrix ops overlap with 6 in-flight global_load_b128 pairs.
// ---------------------------------------------------------------------------
__device__ __forceinline__ void gemm_core(const _Float16* __restrict__ A,
                                          const _Float16* __restrict__ W,
                                          int m0, int n0, int lane,
                                          v8f acc[8]) {
  v16h a0 = load_frag(A, EMB, m0,      0, lane);
  v16h a1 = load_frag(A, EMB, m0 + 16, 0, lane);
  v16h b0 = load_frag(W, EMB, n0,      0, lane);
  v16h b1 = load_frag(W, EMB, n0 + 16, 0, lane);
  v16h b2 = load_frag(W, EMB, n0 + 32, 0, lane);
  v16h b3 = load_frag(W, EMB, n0 + 48, 0, lane);
  for (int k0 = 0; k0 < EMB - 32; k0 += 32) {
    const int kn = k0 + 32;
    v16h na0 = load_frag(A, EMB, m0,      kn, lane);
    v16h na1 = load_frag(A, EMB, m0 + 16, kn, lane);
    v16h nb0 = load_frag(W, EMB, n0,      kn, lane);
    v16h nb1 = load_frag(W, EMB, n0 + 16, kn, lane);
    v16h nb2 = load_frag(W, EMB, n0 + 32, kn, lane);
    v16h nb3 = load_frag(W, EMB, n0 + 48, kn, lane);
    acc[0] = wmma_f16(a0, b0, acc[0]);
    acc[1] = wmma_f16(a0, b1, acc[1]);
    acc[2] = wmma_f16(a0, b2, acc[2]);
    acc[3] = wmma_f16(a0, b3, acc[3]);
    acc[4] = wmma_f16(a1, b0, acc[4]);
    acc[5] = wmma_f16(a1, b1, acc[5]);
    acc[6] = wmma_f16(a1, b2, acc[6]);
    acc[7] = wmma_f16(a1, b3, acc[7]);
    a0 = na0; a1 = na1;
    b0 = nb0; b1 = nb1; b2 = nb2; b3 = nb3;
  }
  acc[0] = wmma_f16(a0, b0, acc[0]);
  acc[1] = wmma_f16(a0, b1, acc[1]);
  acc[2] = wmma_f16(a0, b2, acc[2]);
  acc[3] = wmma_f16(a0, b3, acc[3]);
  acc[4] = wmma_f16(a1, b0, acc[4]);
  acc[5] = wmma_f16(a1, b1, acc[5]);
  acc[6] = wmma_f16(a1, b2, acc[6]);
  acc[7] = wmma_f16(a1, b3, acc[7]);
}

// ---------------------------------------------------------------------------
// Stage 1: C = A @ W^T, f16 out (with scale). grid(MTOT/32, EMB/64), block 32.
// ---------------------------------------------------------------------------
WAVE_KERNEL void gemm_qkv(const _Float16* __restrict__ A,
                          const _Float16* __restrict__ W,
                          _Float16* __restrict__ Out, float scale) {
  const int lane = threadIdx.x;
  const int m0 = blockIdx.x * 32;
  const int n0 = blockIdx.y * 64;
  v8f acc[8] = {};
  gemm_core(A, W, m0, n0, lane, acc);
  const int rb = m0 + 8 * (lane >> 4);
  const int cb = n0 + (lane & 15);
#pragma unroll
  for (int r = 0; r < 8; ++r) {
    size_t row0 = (size_t)(rb + r) * EMB;
    size_t row1 = (size_t)(rb + 16 + r) * EMB;
    Out[row0 + cb +  0] = (_Float16)(acc[0][r] * scale);
    Out[row0 + cb + 16] = (_Float16)(acc[1][r] * scale);
    Out[row0 + cb + 32] = (_Float16)(acc[2][r] * scale);
    Out[row0 + cb + 48] = (_Float16)(acc[3][r] * scale);
    Out[row1 + cb +  0] = (_Float16)(acc[4][r] * scale);
    Out[row1 + cb + 16] = (_Float16)(acc[5][r] * scale);
    Out[row1 + cb + 32] = (_Float16)(acc[6][r] * scale);
    Out[row1 + cb + 48] = (_Float16)(acc[7][r] * scale);
  }
}

// ---------------------------------------------------------------------------
// Stage 2: V[b,s,h,d] -> Vt[b,h,d,s]  (32x32 LDS tile transpose)
// ---------------------------------------------------------------------------
__global__ void transpose_v(const _Float16* __restrict__ Vh,
                            _Float16* __restrict__ Vt) {
  __shared__ _Float16 t[32][33];
  const int bh = blockIdx.z;
  const int b  = bh >> 4;
  const int h  = bh & 15;
  const int s0 = blockIdx.x * 32;
  const int d0 = blockIdx.y * 32;
  const int tx = threadIdx.x, ty = threadIdx.y;
#pragma unroll
  for (int i = 0; i < 32; i += 8)
    t[ty + i][tx] =
        Vh[(size_t)(b * SEQ + s0 + ty + i) * EMB + h * HDIM + d0 + tx];
  __syncthreads();
#pragma unroll
  for (int i = 0; i < 32; i += 8)
    Vt[(size_t)(bh * HDIM + d0 + ty + i) * SEQ + s0 + tx] = t[tx][ty + i];
}

// ---------------------------------------------------------------------------
// Stage 3: flash attention. One wave per (b, h, 16-query tile).
// Score scale (1/32) pre-folded into Q. Pipelined key loop: after the score
// WMMAs consume the K fragments, the registers are immediately refilled with
// the next key tile; V loads issue before the VALU softmax so both overlap.
// ---------------------------------------------------------------------------
#define PSTRIDE 40  // 80B rows: 16B-aligned chunks, conflict-free bank walk

WAVE_KERNEL void attn(const _Float16* __restrict__ Qh,
                      const _Float16* __restrict__ Kh,
                      const _Float16* __restrict__ Vt,
                      _Float16* __restrict__ Oh) {
  __shared__ _Float16 Pl[16 * PSTRIDE];
  const int lane = threadIdx.x;
  const int h = blockIdx.y, b = blockIdx.z;
  const int hd = h * HDIM;
  const int q0 = b * SEQ + blockIdx.x * 16;
  const int krow = b * SEQ;
  const int vrow = (b * HEADS + h) * HDIM;

  const v16h qlo = load_frag(Qh, EMB, q0, hd,      lane);
  const v16h qhi = load_frag(Qh, EMB, q0, hd + 32, lane);

  v8f acc0 = {}, acc1 = {}, acc2 = {}, acc3 = {};
  float m_i[8], l_i[8];
#pragma unroll
  for (int r = 0; r < 8; ++r) { m_i[r] = -1e30f; l_i[r] = 0.0f; }

  const int pr = 8 * (lane >> 4);
  const int pc = lane & 15;

  // prologue: K fragments for kt = 0
  v16h klo0 = load_frag(Kh, EMB, krow +  0, hd,      lane);
  v16h khi0 = load_frag(Kh, EMB, krow +  0, hd + 32, lane);
  v16h klo1 = load_frag(Kh, EMB, krow + 16, hd,      lane);
  v16h khi1 = load_frag(Kh, EMB, krow + 16, hd + 32, lane);

  for (int kt = 0; kt < SEQ; kt += 32) {
    // ---- scores: two 16q x 16k tiles, K-dim = 64 ----
    v8f s0 = {}, s1 = {};
    s0 = wmma_f16(qlo, klo0, s0);
    s0 = wmma_f16(qhi, khi0, s0);
    s1 = wmma_f16(qlo, klo1, s1);
    s1 = wmma_f16(qhi, khi1, s1);

    // ---- refill K registers with the NEXT key tile (overlaps softmax) ----
    if (kt + 32 < SEQ) {
      klo0 = load_frag(Kh, EMB, krow + kt + 32, hd,      lane);
      khi0 = load_frag(Kh, EMB, krow + kt + 32, hd + 32, lane);
      klo1 = load_frag(Kh, EMB, krow + kt + 48, hd,      lane);
      khi1 = load_frag(Kh, EMB, krow + kt + 48, hd + 32, lane);
      __builtin_prefetch(Vt + (size_t)(vrow + (lane & 15)) * SEQ + kt + 32,
                         0, 1);
    }
    // ---- V fragments for the CURRENT tile (independent of softmax) ----
    v16h bv0 = load_frag(Vt, SEQ, vrow +  0, kt, lane);
    v16h bv1 = load_frag(Vt, SEQ, vrow + 16, kt, lane);
    v16h bv2 = load_frag(Vt, SEQ, vrow + 32, kt, lane);
    v16h bv3 = load_frag(Vt, SEQ, vrow + 48, kt, lane);

    // ---- online softmax over the 32 new columns ----
    float alpha[8];
#pragma unroll
    for (int r = 0; r < 8; ++r) {
      float mx = rmax16(fmaxf(s0[r], s1[r]));
      float mnew = fmaxf(m_i[r], mx);
      alpha[r] = __expf(m_i[r] - mnew);
      float p0 = __expf(s0[r] - mnew);
      float p1 = __expf(s1[r] - mnew);
      s0[r] = p0; s1[r] = p1;
      l_i[r] = l_i[r] * alpha[r] + rsum16(p0 + p1);
      m_i[r] = mnew;
    }
#pragma unroll
    for (int r = 0; r < 8; ++r) {
      acc0[r] *= alpha[r]; acc1[r] *= alpha[r];
      acc2[r] *= alpha[r]; acc3[r] *= alpha[r];
    }
    // ---- stage P (16x32 f16) through LDS to re-layout as an A-fragment ----
#pragma unroll
    for (int r = 0; r < 8; ++r) {
      Pl[(pr + r) * PSTRIDE + pc]      = (_Float16)s0[r];
      Pl[(pr + r) * PSTRIDE + pc + 16] = (_Float16)s1[r];
    }
    asm volatile("s_wait_dscnt 0" ::: "memory");
    v16h pa = load_frag(Pl, PSTRIDE, 0, 0, lane);
    // (no second wait needed: same-wave DS ops are processed in order, so the
    //  next iteration's ds_stores cannot bypass these ds_loads)

    // ---- O += P @ V ----
    acc0 = wmma_f16(pa, bv0, acc0);
    acc1 = wmma_f16(pa, bv1, acc1);
    acc2 = wmma_f16(pa, bv2, acc2);
    acc3 = wmma_f16(pa, bv3, acc3);
  }
  // ---- normalize and store O[b,s,h,d] as f16 ----
#pragma unroll
  for (int r = 0; r < 8; ++r) {
    float inv = 1.0f / l_i[r];
    size_t row = (size_t)(q0 + pr + r) * EMB + hd + pc;
    Oh[row +  0] = (_Float16)(acc0[r] * inv);
    Oh[row + 16] = (_Float16)(acc1[r] * inv);
    Oh[row + 32] = (_Float16)(acc2[r] * inv);
    Oh[row + 48] = (_Float16)(acc3[r] * inv);
  }
}

// ---------------------------------------------------------------------------
// Stage 4: out = O @ Wo^T + bo, f32 output. grid(MTOT/32, EMB/64), block 32.
// ---------------------------------------------------------------------------
WAVE_KERNEL void gemm_out(const _Float16* __restrict__ A,
                          const _Float16* __restrict__ W,
                          const float* __restrict__ bias,
                          float* __restrict__ Out) {
  const int lane = threadIdx.x;
  const int m0 = blockIdx.x * 32;
  const int n0 = blockIdx.y * 64;
  v8f acc[8] = {};
  gemm_core(A, W, m0, n0, lane, acc);
  const int rb = m0 + 8 * (lane >> 4);
  const int cb = n0 + (lane & 15);
  float bi0 = bias[cb], bi1 = bias[cb + 16];
  float bi2 = bias[cb + 32], bi3 = bias[cb + 48];
#pragma unroll
  for (int r = 0; r < 8; ++r) {
    size_t row0 = (size_t)(rb + r) * EMB;
    size_t row1 = (size_t)(rb + 16 + r) * EMB;
    Out[row0 + cb +  0] = acc[0][r] + bi0;
    Out[row0 + cb + 16] = acc[1][r] + bi1;
    Out[row0 + cb + 32] = acc[2][r] + bi2;
    Out[row0 + cb + 48] = acc[3][r] + bi3;
    Out[row1 + cb +  0] = acc[4][r] + bi0;
    Out[row1 + cb + 16] = acc[5][r] + bi1;
    Out[row1 + cb + 32] = acc[6][r] + bi2;
    Out[row1 + cb + 48] = acc[7][r] + bi3;
  }
}

// ---------------------------------------------------------------------------
extern "C" void kernel_launch(void* const* d_in, const int* in_sizes, int n_in,
                              void* d_out, int out_size, void* d_ws, size_t ws_size,
                              hipStream_t stream) {
  (void)in_sizes; (void)n_in; (void)out_size; (void)ws_size;
  const float* x  = (const float*)d_in[0];
  const float* Wk = (const float*)d_in[1];
  const float* Wq = (const float*)d_in[2];
  const float* Wv = (const float*)d_in[3];
  const float* Wo = (const float*)d_in[4];
  const float* bo = (const float*)d_in[5];
  float* out = (float*)d_out;

  _Float16* w = (_Float16*)d_ws;
  const size_t NX = (size_t)MTOT * EMB;     // 4,194,304
  const size_t NW = (size_t)EMB * EMB;      // 1,048,576
  _Float16* Xh  = w;
  _Float16* Wkh = Xh  + NX;
  _Float16* Wqh = Wkh + NW;
  _Float16* Wvh = Wqh + NW;
  _Float16* Woh = Wvh + NW;
  _Float16* Qh  = Woh + NW;
  _Float16* Kh  = Qh  + NX;
  _Float16* Vh  = Kh  + NX;
  _Float16* Vt  = Vh  + NX;
  _Float16* Oh  = Vt  + NX;   // total ~56 MB of workspace

  // Stage 0: conversions
  {
    int n4 = (int)(NX / 4);
    cvt_f32_f16<<<dim3((n4 + 255) / 256), dim3(256), 0, stream>>>(x, Xh, n4);
    int w4 = (int)(NW / 4);
    dim3 g((w4 + 255) / 256), blk(256);
    cvt_f32_f16<<<g, blk, 0, stream>>>(Wk, Wkh, w4);
    cvt_f32_f16<<<g, blk, 0, stream>>>(Wq, Wqh, w4);
    cvt_f32_f16<<<g, blk, 0, stream>>>(Wv, Wvh, w4);
    cvt_f32_f16<<<g, blk, 0, stream>>>(Wo, Woh, w4);
  }

  // Stage 1: Q/K/V projections (score scale folded into Q)
  {
    dim3 grid(MTOT / 32, EMB / 64), blk(32);
    gemm_qkv<<<grid, blk, 0, stream>>>(Xh, Wqh, Qh, 1.0f / 32.0f);
    gemm_qkv<<<grid, blk, 0, stream>>>(Xh, Wkh, Kh, 1.0f);
    gemm_qkv<<<grid, blk, 0, stream>>>(Xh, Wvh, Vh, 1.0f);
  }

  // Stage 2: V transpose to [b,h,d,s]
  transpose_v<<<dim3(SEQ / 32, HDIM / 32, BATCH * HEADS), dim3(32, 8), 0,
                stream>>>(Vh, Vt);

  // Stage 3: attention
  attn<<<dim3(SEQ / 16, HEADS, BATCH), dim3(32), 0, stream>>>(Qh, Kh, Vt, Oh);

  // Stage 4: output projection + bias
  gemm_out<<<dim3(MTOT / 32, EMB / 64), dim3(32), 0, stream>>>(Oh, Woh, bo, out);
}
